// featureGenerationNet2_13297218748540
// MI455X (gfx1250) — matrix operand order repaired and checked by reading, not compiled, not run on hardware
//
#include <hip/hip_runtime.h>

// ---------------------------------------------------------------------------
// featureGenerationNet2: 4x GraphConv (segment_sum + dense) + fused WMMA MLP
// MI455X / gfx1250, wave32.
// ---------------------------------------------------------------------------

typedef __attribute__((ext_vector_type(16))) _Float16 v16h;
typedef __attribute__((ext_vector_type(8)))  float    v8f;

#define N_NODES 500000
#define N_EDGES 16000000

// ---------------------------------------------------------------------------
// Zero-fill (agg buffer re-init each layer; graph-capture safe, deterministic)
// ---------------------------------------------------------------------------
__global__ __launch_bounds__(256) void zero_kernel(float* __restrict__ p, int n) {
  int i = blockIdx.x * 256 + threadIdx.x;
  int stride = gridDim.x * 256;
  for (; i < n; i += stride) p[i] = 0.0f;
}

// ---------------------------------------------------------------------------
// Edge scatter: agg[dst] += x[src].  Index streams (512 MB over 4 layers) are
// marked non-temporal so they stream from HBM without evicting the L2-resident
// node-feature arrays (<= 20 MB) that the random gathers/atomics hit.
// ---------------------------------------------------------------------------
template<int DIN>
__global__ __launch_bounds__(256) void edge_scatter_kernel(
    const int* __restrict__ src, const int* __restrict__ dst,
    const float* __restrict__ x, float* __restrict__ agg)
{
  int e = blockIdx.x * 256 + threadIdx.x;
  if (e >= N_EDGES) return;
  int s = __builtin_nontemporal_load(src + e);
  int d = __builtin_nontemporal_load(dst + e);
  const float* xs = x + (size_t)s * DIN;
  float*       ad = agg + (size_t)d * DIN;
#pragma unroll
  for (int k = 0; k < DIN; ++k) atomicAdd(&ad[k], xs[k]);
}

// ---------------------------------------------------------------------------
// Per-node GraphConv transform: out = relu(agg @ Wrel + brel + x @ Wroot)
// Weights are uniform across lanes -> scalar loads; din*dout <= 160 FMAs.
// ---------------------------------------------------------------------------
template<int DIN, int DOUT>
__global__ __launch_bounds__(256) void node_update_kernel(
    const float* __restrict__ x, const float* __restrict__ agg,
    const float* __restrict__ Wrel, const float* __restrict__ brel,
    const float* __restrict__ Wroot, float* __restrict__ out)
{
  int i = blockIdx.x * 256 + threadIdx.x;
  if (i >= N_NODES) return;
  float xi[DIN], ai[DIN];
#pragma unroll
  for (int k = 0; k < DIN; ++k) {
    xi[k] = x[(size_t)i * DIN + k];
    ai[k] = agg[(size_t)i * DIN + k];
  }
#pragma unroll
  for (int o = 0; o < DOUT; ++o) {
    float acc = brel[o];
#pragma unroll
    for (int k = 0; k < DIN; ++k)
      acc = fmaf(ai[k], Wrel[k * DOUT + o], fmaf(xi[k], Wroot[k * DOUT + o], acc));
    out[(size_t)i * DOUT + o] = fmaxf(acc, 0.0f);
  }
}

// ---------------------------------------------------------------------------
// Fused MLP via WMMA f32_16x16x32_f16.  One wave owns a 16-node tile; fc1/fc2
// activations live in a per-wave LDS slice (same-wave LDS ops are in-order ->
// no barriers).  K=16 stages are zero-padded to K=32.
// ---------------------------------------------------------------------------
__device__ __forceinline__ v8f wmma16(v16h a, v16h b, v8f c) {
  // 8 args: (neg_a, A, neg_b, B, c_mod, C, reuse_a, reuse_b)
  return __builtin_amdgcn_wmma_f32_16x16x32_f16(false, a, false, b, (short)0, c,
                                                false, false);
}

// A-matrix (16x32 f16) per ISA table: lane l -> row M=l&15, group g=l>>4;
// VGPR v holds K = 2v+8g (v<4) or 16+2(v-4)+8g (v>=4).
__device__ __forceinline__ v16h make_A(const float (*H)[33], int lane, int K) {
  int m = lane & 15, g = lane >> 4;
  v16h a;
#pragma unroll
  for (int v = 0; v < 8; ++v) {
    int k0 = (v < 4) ? (2 * v + 8 * g) : (16 + 2 * (v - 4) + 8 * g);
    a[2 * v]     = (k0     < K) ? (_Float16)H[m][k0]     : (_Float16)0.0f;
    a[2 * v + 1] = (k0 + 1 < K) ? (_Float16)H[m][k0 + 1] : (_Float16)0.0f;
  }
  return a;
}

// B-matrix (32x16 f16): lane l -> col N=l&15, group g=l>>4; VGPR v holds
// K = 2v+8g (v<4) or 16+2(v-4)+8g (v>=4).  W is row-major [K x ldw], columns [ncol0, ncol0+16).
__device__ __forceinline__ v16h make_B(const float* __restrict__ W, int ldw,
                                       int ncol0, int K, int lane) {
  int n = lane & 15, g = lane >> 4;
  v16h b;
#pragma unroll
  for (int v = 0; v < 8; ++v) {
    int k0 = (v < 4) ? (2 * v + 8 * g) : (16 + 2 * (v - 4) + 8 * g);
    b[2 * v]     = (k0     < K) ? (_Float16)W[k0 * ldw + ncol0 + n]       : (_Float16)0.0f;
    b[2 * v + 1] = (k0 + 1 < K) ? (_Float16)W[(k0 + 1) * ldw + ncol0 + n] : (_Float16)0.0f;
  }
  return b;
}

// C/D (16x16 f32): VGPR r -> M = r+8g, N = lane&15.  Bias + optional relu,
// written back to the LDS tile for the next stage's A.
__device__ __forceinline__ void store_C(float (*H)[33], v8f c,
                                        const float* __restrict__ bias,
                                        int ncol0, int lane, bool relu) {
  int n = lane & 15, g = lane >> 4;
  float bv = bias[ncol0 + n];
#pragma unroll
  for (int r = 0; r < 8; ++r) {
    float val = c[r] + bv;
    if (relu) val = fmaxf(val, 0.0f);
    H[r + 8 * g][ncol0 + n] = val;
  }
}

__global__ __launch_bounds__(256) void mlp_fused_kernel(
    const float* __restrict__ h4,                        // [N,16]
    const float* __restrict__ fc1W, const float* __restrict__ fc1b,  // 16x32, 32
    const float* __restrict__ fc2W, const float* __restrict__ fc2b,  // 32x16, 16
    const float* __restrict__ fc3W, const float* __restrict__ fc3b,  // 16x16, 16
    float* __restrict__ out, int nTiles)                 // [N,16]
{
  __shared__ float lds[8][16][33];  // per-wave 16x32 tile (stride 33: no bank conflicts)
  int wave = threadIdx.x >> 5;
  int lane = threadIdx.x & 31;
  int t = blockIdx.x * 8 + wave;
  if (t >= nTiles) return;  // wave-uniform -> EXEC all-ones for WMMA
  float (*H)[33] = lds[wave];

  int m = lane & 15, g = lane >> 4;

  // Stage input tile (16 rows x 16 f32) into LDS: lane loads row m, 8 cols.
  {
    const float4* s4 = (const float4*)(h4 + (size_t)(t * 16 + m) * 16);
    float4 v0 = s4[g * 2 + 0];
    float4 v1 = s4[g * 2 + 1];
    H[m][g * 8 + 0] = v0.x; H[m][g * 8 + 1] = v0.y;
    H[m][g * 8 + 2] = v0.z; H[m][g * 8 + 3] = v0.w;
    H[m][g * 8 + 4] = v1.x; H[m][g * 8 + 5] = v1.y;
    H[m][g * 8 + 6] = v1.z; H[m][g * 8 + 7] = v1.w;
  }

  v8f zero = {};

  // fc1: [16x16] @ [16x32] -> 16x32 (two N-halves), K padded 16->32
  {
    v16h a  = make_A(H, lane, 16);
    v16h bl = make_B(fc1W, 32, 0, 16, lane);
    v16h bh = make_B(fc1W, 32, 16, 16, lane);
    v8f cl = wmma16(a, bl, zero);
    v8f ch = wmma16(a, bh, zero);
    store_C(H, cl, fc1b, 0, lane, true);
    store_C(H, ch, fc1b, 16, lane, true);
  }
  // fc2: [16x32] @ [32x16] -> 16x16
  {
    v16h a = make_A(H, lane, 32);
    v16h b = make_B(fc2W, 16, 0, 32, lane);
    v8f c = wmma16(a, b, zero);
    store_C(H, c, fc2b, 0, lane, true);
  }
  // fc3: [16x16] @ [16x16] -> 16x16, bias, no relu, straight to global
  {
    v16h a = make_A(H, lane, 16);
    v16h b = make_B(fc3W, 16, 0, 16, lane);
    v8f c = wmma16(a, b, zero);
    int n = lane & 15;
    float bv = fc3b[n];
#pragma unroll
    for (int r = 0; r < 8; ++r)
      out[(size_t)(t * 16 + r + 8 * g) * 16 + n] = c[r] + bv;
  }
}

// ---------------------------------------------------------------------------
// Host launcher
// ---------------------------------------------------------------------------
extern "C" void kernel_launch(void* const* d_in, const int* in_sizes, int n_in,
                              void* d_out, int out_size, void* d_ws, size_t ws_size,
                              hipStream_t stream) {
  const float* x    = (const float*)d_in[0];
  const int*   ei   = (const int*)d_in[1];
  const int*   src  = ei;             // edge_index[0, :]
  const int*   dst  = ei + N_EDGES;   // edge_index[1, :]
  const float* g1Wrel = (const float*)d_in[2],  *g1brel = (const float*)d_in[3],  *g1Wroot = (const float*)d_in[4];
  const float* g2Wrel = (const float*)d_in[5],  *g2brel = (const float*)d_in[6],  *g2Wroot = (const float*)d_in[7];
  const float* g3Wrel = (const float*)d_in[8],  *g3brel = (const float*)d_in[9],  *g3Wroot = (const float*)d_in[10];
  const float* g4Wrel = (const float*)d_in[11], *g4brel = (const float*)d_in[12], *g4Wroot = (const float*)d_in[13];
  const float* fc1W = (const float*)d_in[14], *fc1b = (const float*)d_in[15];
  const float* fc2W = (const float*)d_in[16], *fc2b = (const float*)d_in[17];
  const float* fc3W = (const float*)d_in[18], *fc3b = (const float*)d_in[19];
  float* out = (float*)d_out;

  // Workspace: agg (5M f32) | bufA (5M f32) | bufB (8M f32)  == 72 MB
  float* ws   = (float*)d_ws;
  float* agg  = ws;
  float* bufA = ws + 5000000;
  float* bufB = bufA + 5000000;

  const int edgeBlocks = N_EDGES / 256;            // 62500 exact
  const int nodeBlocks = (N_NODES + 255) / 256;    // 1954
  const int zeroBlocks = 4096;

  // Layer 1: 1 -> 4
  zero_kernel<<<zeroBlocks, 256, 0, stream>>>(agg, N_NODES * 1);
  edge_scatter_kernel<1><<<edgeBlocks, 256, 0, stream>>>(src, dst, x, agg);
  node_update_kernel<1, 4><<<nodeBlocks, 256, 0, stream>>>(x, agg, g1Wrel, g1brel, g1Wroot, bufA);
  // Layer 2: 4 -> 7
  zero_kernel<<<zeroBlocks, 256, 0, stream>>>(agg, N_NODES * 4);
  edge_scatter_kernel<4><<<edgeBlocks, 256, 0, stream>>>(src, dst, bufA, agg);
  node_update_kernel<4, 7><<<nodeBlocks, 256, 0, stream>>>(bufA, agg, g2Wrel, g2brel, g2Wroot, bufB);
  // Layer 3: 7 -> 10
  zero_kernel<<<zeroBlocks, 256, 0, stream>>>(agg, N_NODES * 7);
  edge_scatter_kernel<7><<<edgeBlocks, 256, 0, stream>>>(src, dst, bufB, agg);
  node_update_kernel<7, 10><<<nodeBlocks, 256, 0, stream>>>(bufB, agg, g3Wrel, g3brel, g3Wroot, bufA);
  // Layer 4: 10 -> 16
  zero_kernel<<<zeroBlocks, 256, 0, stream>>>(agg, N_NODES * 10);
  edge_scatter_kernel<10><<<edgeBlocks, 256, 0, stream>>>(src, dst, bufA, agg);
  node_update_kernel<10, 16><<<nodeBlocks, 256, 0, stream>>>(bufA, agg, g4Wrel, g4brel, g4Wroot, bufB);

  // Fused MLP (WMMA): 31250 tiles of 16 nodes, 8 waves/block
  const int nTiles = N_NODES / 16;                 // 31250 exact
  mlp_fused_kernel<<<(nTiles + 7) / 8, 256, 0, stream>>>(
      bufB, fc1W, fc1b, fc2W, fc2b, fc3W, fc3b, out, nTiles);
}
